// MultiHeadAttention_33569464386050
// MI455X (gfx1250) — compile-verified, matrix operand order checked by reading, not run on hardware
//
#include <hip/hip_runtime.h>

// ---------------------------------------------------------------------------
// MHA for MI455X (gfx1250, wave32): bf16 WMMA, f32 accumulation.
// B=4, L=2048, D=1024, H=16, DH=64.
// GEMM stages the shared W^T panel into LDS with async-to-LDS copies
// (ASYNCcnt path); all WMMA fragments are contiguous dword loads.
// ---------------------------------------------------------------------------

#define DEV __device__ __forceinline__

constexpr int Bc  = 4;
constexpr int Lc  = 2048;
constexpr int Dc  = 1024;
constexpr int Hc  = 16;
constexpr int DHc = 64;

typedef __attribute__((ext_vector_type(16))) __bf16 v16bf;
typedef __attribute__((ext_vector_type(8)))  float  v8f;

union FragBF { v16bf v; unsigned u[8]; };

DEV unsigned short f2bf(float f) {
  unsigned u = __builtin_bit_cast(unsigned, f);
  return (unsigned short)((u + 0x7FFFu + ((u >> 16) & 1u)) >> 16);
}

DEV v8f wmma_bf16(const FragBF& a, const FragBF& b, v8f c) {
  // (neg_a, A, neg_b, B, c_mod, C, reuse_a, reuse_b)
  return __builtin_amdgcn_wmma_f32_16x16x32_bf16(false, a.v, false, b.v,
                                                 (short)0, c, false, false);
}

// 16-bit A/B fragment K-offset for VGPR v, lane-half hi8 (0 or 8):
// VGPR0..3 -> K = hi8 + 2*(v&3) ; VGPR4..7 -> K = 16 + hi8 + 2*(v&3)
DEV int kpat(int v, int hi8) { return ((v & 4) << 2) + hi8 + ((v & 3) << 1); }

// Async copy of 16 bytes global -> LDS (ASYNCcnt-tracked, no VGPR data).
DEV void async_copy_b128(const void* gptr, unsigned ldsByteOff) {
  asm volatile("global_load_async_to_lds_b128 %0, %1, off"
               :: "v"(ldsByteOff),
                  "v"((unsigned long long)(size_t)gptr)
               : "memory");
}
DEV void wait_async_then_barrier() {
  asm volatile("s_wait_asynccnt 0" ::: "memory");
  __syncthreads();
}

// ---------------------------------------------------------------------------
// f32 -> bf16 cast (vectorized by 4)
// ---------------------------------------------------------------------------
__global__ void cast_bf16_kernel(const float* __restrict__ src,
                                 unsigned short* __restrict__ dst, int n4) {
  int i = blockIdx.x * blockDim.x + threadIdx.x;
  if (i < n4) {
    float4 f = reinterpret_cast<const float4*>(src)[i];
    ushort4 o;
    o.x = f2bf(f.x); o.y = f2bf(f.y); o.z = f2bf(f.z); o.w = f2bf(f.w);
    reinterpret_cast<ushort4*>(dst)[i] = o;
  }
}

// ---------------------------------------------------------------------------
// f32 [R][C] -> bf16 transposed [C][R], 32x32 LDS tiles. R,C multiples of 32.
// ---------------------------------------------------------------------------
__global__ __launch_bounds__(256) void castT_bf16_kernel(
    const float* __restrict__ src, unsigned short* __restrict__ dst,
    int R, int C) {
  __shared__ unsigned short t[32][33];
  const int c0 = blockIdx.x * 32;
  const int r0 = blockIdx.y * 32;
  for (int i = threadIdx.y; i < 32; i += 8)
    t[i][threadIdx.x] = f2bf(src[(size_t)(r0 + i) * C + c0 + threadIdx.x]);
  __syncthreads();
  for (int i = threadIdx.y; i < 32; i += 8)
    dst[(size_t)(c0 + i) * R + r0 + threadIdx.x] = t[threadIdx.x][i];
}

// ---------------------------------------------------------------------------
// bf16 per-batch transpose: src [BATCH][R][C] -> dst [BATCH][C][R]
// ---------------------------------------------------------------------------
__global__ __launch_bounds__(256) void transpose_bf16_kernel(
    const unsigned short* __restrict__ src, unsigned short* __restrict__ dst,
    int R, int C) {
  __shared__ unsigned short t[32][33];
  const int c0 = blockIdx.x * 32;
  const int r0 = blockIdx.y * 32;
  const size_t sb = (size_t)blockIdx.z * R * C;
  for (int i = threadIdx.y; i < 32; i += 8)
    t[i][threadIdx.x] = src[sb + (size_t)(r0 + i) * C + c0 + threadIdx.x];
  __syncthreads();
  for (int i = threadIdx.y; i < 32; i += 8)
    dst[sb + (size_t)(c0 + i) * R + r0 + threadIdx.x] = t[threadIdx.x][i];
}

// ---------------------------------------------------------------------------
// GEMM: C[M,N] = A[M,K](bf16) @ W[K,N] + bias[N], W given TRANSPOSED
// (WT[N][K], bf16). Block = 8 waves = 256 rows x 64 cols; the 64xK W^T panel
// is staged through LDS in 64x256 chunks with async-to-LDS copies shared by
// all 8 waves. Each wave: 32x64 strip, 8 WMMA per 32-wide K-step.
// Requires: M % 256 == 0, N % 64 == 0, K % 256 == 0.
// ---------------------------------------------------------------------------
__global__ __launch_bounds__(256) void gemm_bias_kernel(
    const unsigned short* __restrict__ A,
    const unsigned short* __restrict__ WT,
    const float* __restrict__ bias,
    unsigned short* __restrict__ outBF,
    float* __restrict__ outF,
    int M, int N, int K) {
  __shared__ unsigned short Bpanel[64 * 256];   // 32 KB chunk of W^T panel

  const int tid  = threadIdx.x;
  const int lane = tid & 31;
  const int wave = tid >> 5;
  const int li   = lane & 15;
  const int hi8  = (lane >> 4) << 3;

  const int ntiles = N >> 6;
  const int nt   = blockIdx.x % ntiles;
  const int mrow = ((blockIdx.x / ntiles) * 8 + wave) * 32;

  const unsigned short* Arow0  = A + (size_t)(mrow + li) * K;
  const unsigned short* Arow1  = Arow0 + (size_t)16 * K;
  const unsigned short* Wpanel = WT + (size_t)nt * 64 * K;

  v8f c[8] = {v8f{}, v8f{}, v8f{}, v8f{}, v8f{}, v8f{}, v8f{}, v8f{}};

  for (int kc = 0; kc < K; kc += 256) {
    // ---- cooperative async stage of 64x256 chunk (2048 x b128) ----
#pragma unroll
    for (int it = 0; it < 8; ++it) {
      int id    = it * 256 + tid;
      int row   = id >> 5;        // 0..63
      int col16 = id & 31;        // 16B segment within 256-elem row
      const unsigned short* g = Wpanel + (size_t)row * K + kc + col16 * 8;
      unsigned ldsOff = (unsigned)(size_t)&Bpanel[row * 256 + col16 * 8];
      async_copy_b128(g, ldsOff);
    }
    wait_async_then_barrier();

    // ---- consume chunk: 8 K-steps x 8 WMMA ----
    for (int k0 = 0; k0 < 256; k0 += 32) {
      FragBF a0, a1;
#pragma unroll
      for (int v = 0; v < 8; ++v) {
        int kp = kc + k0 + kpat(v, hi8);
        a0.u[v] = *reinterpret_cast<const unsigned*>(Arow0 + kp);
        a1.u[v] = *reinterpret_cast<const unsigned*>(Arow1 + kp);
      }
#pragma unroll
      for (int j = 0; j < 4; ++j) {
        const unsigned short* bn = &Bpanel[(j * 16 + li) * 256 + k0];
        FragBF bf;
#pragma unroll
        for (int v = 0; v < 8; ++v)
          bf.u[v] = *reinterpret_cast<const unsigned*>(bn + kpat(v, hi8));
        c[j]     = wmma_bf16(a0, bf, c[j]);
        c[4 + j] = wmma_bf16(a1, bf, c[4 + j]);
      }
    }
    __syncthreads();   // protect LDS before next chunk overwrites it
  }

  const int n0 = nt * 64 + li;
  const float bs0 = bias[n0], bs1 = bias[n0 + 16];
  const float bs2 = bias[n0 + 32], bs3 = bias[n0 + 48];
#pragma unroll
  for (int half = 0; half < 2; ++half) {
#pragma unroll
    for (int r = 0; r < 8; ++r) {
      const size_t base = (size_t)(mrow + half * 16 + r + hi8) * N;
      float o0 = c[half * 4 + 0][r] + bs0;
      float o1 = c[half * 4 + 1][r] + bs1;
      float o2 = c[half * 4 + 2][r] + bs2;
      float o3 = c[half * 4 + 3][r] + bs3;
      if (outF) {
        outF[base + n0]      = o0; outF[base + n0 + 16] = o1;
        outF[base + n0 + 32] = o2; outF[base + n0 + 48] = o3;
      } else {
        outBF[base + n0]      = f2bf(o0); outBF[base + n0 + 16] = f2bf(o1);
        outBF[base + n0 + 32] = f2bf(o2); outBF[base + n0 + 48] = f2bf(o3);
      }
    }
  }
}

// ---------------------------------------------------------------------------
// Flash attention: one wave per (b, h, 16-query tile). 8 waves / block.
// Per 32-key step: 4 WMMA (Q.K^T) + online softmax + 4 WMMA (P.V).
// V is consumed pre-transposed: Vt[b][d][l]  (d = h*DH + dh).
// P tile round-trips through per-wave LDS to convert C-layout -> A-layout.
// ---------------------------------------------------------------------------
__global__ __launch_bounds__(256) void attn_kernel(
    const unsigned short* __restrict__ Q,
    const unsigned short* __restrict__ Km,
    const unsigned short* __restrict__ Vt,
    unsigned short* __restrict__ Ctx) {
  __shared__ unsigned short Pbuf[8][16 * 32];

  const int lane = threadIdx.x & 31;
  const int wave = threadIdx.x >> 5;
  const int li   = lane & 15;
  const int hi8  = (lane >> 4) << 3;

  const int wid = blockIdx.x * 8 + wave;           // [0, B*H*L/16)
  const int qt  = wid % (Lc / 16);
  const int bh  = wid / (Lc / 16);
  const int h   = bh % Hc;
  const int b   = bh / Hc;

  // Q tile A-fragments: 16 rows x 64 features -> two 16x32 frags
  const size_t qrow = ((size_t)b * Lc + qt * 16 + li) * Dc + h * DHc;
  FragBF qa0, qa1;
#pragma unroll
  for (int v = 0; v < 8; ++v) {
    int kp = kpat(v, hi8);
    qa0.u[v] = *reinterpret_cast<const unsigned*>(Q + qrow + kp);
    qa1.u[v] = *reinterpret_cast<const unsigned*>(Q + qrow + 32 + kp);
  }

  float mrow[8], lrow[8];
#pragma unroll
  for (int r = 0; r < 8; ++r) { mrow[r] = -1e30f; lrow[r] = 0.0f; }
  v8f acc[4] = {v8f{}, v8f{}, v8f{}, v8f{}};
  const float scale = 0.125f;                       // 1/sqrt(64)

  unsigned short* pb = Pbuf[wave];
  const unsigned short* VtHead = Vt + ((size_t)b * Dc + h * DHc) * Lc;

  for (int kv = 0; kv < Lc; kv += 32) {
    // ---- scores S = Q . K^T for keys [kv, kv+32) ----
    const unsigned short* Krow0 = Km + ((size_t)b * Lc + kv + li) * Dc + h * DHc;
    const unsigned short* Krow1 = Krow0 + (size_t)16 * Dc;
    FragBF kb00, kb01, kb10, kb11;
#pragma unroll
    for (int v = 0; v < 8; ++v) {
      int kp = kpat(v, hi8);
      kb00.u[v] = *reinterpret_cast<const unsigned*>(Krow0 + kp);
      kb01.u[v] = *reinterpret_cast<const unsigned*>(Krow0 + 32 + kp);
      kb10.u[v] = *reinterpret_cast<const unsigned*>(Krow1 + kp);
      kb11.u[v] = *reinterpret_cast<const unsigned*>(Krow1 + 32 + kp);
    }
    v8f s0 = {}, s1 = {};
    s0 = wmma_bf16(qa0, kb00, s0);
    s0 = wmma_bf16(qa1, kb01, s0);
    s1 = wmma_bf16(qa0, kb10, s1);
    s1 = wmma_bf16(qa1, kb11, s1);

    // ---- online softmax (row stats across 16-lane column groups) ----
    float p0[8], p1[8], esc[8];
#pragma unroll
    for (int r = 0; r < 8; ++r) {
      float t0 = s0[r] * scale, t1 = s1[r] * scale;
      float mx = fmaxf(t0, t1);
      mx = fmaxf(mx, __shfl_xor(mx, 1, 32));
      mx = fmaxf(mx, __shfl_xor(mx, 2, 32));
      mx = fmaxf(mx, __shfl_xor(mx, 4, 32));
      mx = fmaxf(mx, __shfl_xor(mx, 8, 32));
      float mnew = fmaxf(mrow[r], mx);
      float e0 = __expf(t0 - mnew);
      float e1 = __expf(t1 - mnew);
      float rs = e0 + e1;
      rs += __shfl_xor(rs, 1, 32);
      rs += __shfl_xor(rs, 2, 32);
      rs += __shfl_xor(rs, 4, 32);
      rs += __shfl_xor(rs, 8, 32);
      float es = __expf(mrow[r] - mnew);
      lrow[r] = lrow[r] * es + rs;
      mrow[r] = mnew;
      esc[r] = es;
      p0[r] = e0;
      p1[r] = e1;
    }
#pragma unroll
    for (int j = 0; j < 4; ++j)
#pragma unroll
      for (int r = 0; r < 8; ++r) acc[j][r] *= esc[r];

    // ---- P (C-layout) -> LDS -> A-layout fragment ----
#pragma unroll
    for (int r = 0; r < 8; ++r) {
      int rr = r + hi8;
      pb[rr * 32 + li]      = f2bf(p0[r]);
      pb[rr * 32 + 16 + li] = f2bf(p1[r]);
    }
    __builtin_amdgcn_wave_barrier();
    asm volatile("s_wait_dscnt 0" ::: "memory");
    __builtin_amdgcn_wave_barrier();
    FragBF pf;
#pragma unroll
    for (int v = 0; v < 8; ++v) {
      int kp = kpat(v, hi8);
      pf.u[v] = *reinterpret_cast<const unsigned*>(pb + li * 32 + kp);
    }

    // ---- ctx += P . V  (V transposed: contiguous along keys) ----
#pragma unroll
    for (int j = 0; j < 4; ++j) {
      const unsigned short* vrow = VtHead + (size_t)(j * 16 + li) * Lc + kv;
      FragBF vf;
#pragma unroll
      for (int v = 0; v < 8; ++v)
        vf.u[v] = *reinterpret_cast<const unsigned*>(vrow + kpat(v, hi8));
      acc[j] = wmma_bf16(pf, vf, acc[j]);
    }
  }

  // ---- normalize and write ctx (bf16) ----
#pragma unroll
  for (int r = 0; r < 8; ++r) {
    float inv = 1.0f / lrow[r];
    size_t base = ((size_t)b * Lc + qt * 16 + r + hi8) * Dc + h * DHc;
    Ctx[base + li]      = f2bf(acc[0][r] * inv);
    Ctx[base + 16 + li] = f2bf(acc[1][r] * inv);
    Ctx[base + 32 + li] = f2bf(acc[2][r] * inv);
    Ctx[base + 48 + li] = f2bf(acc[3][r] * inv);
  }
}

// ---------------------------------------------------------------------------
// Host launcher
// ---------------------------------------------------------------------------
extern "C" void kernel_launch(void* const* d_in, const int* in_sizes, int n_in,
                              void* d_out, int out_size, void* d_ws, size_t ws_size,
                              hipStream_t stream) {
  const float* key   = (const float*)d_in[0];
  const float* value = (const float*)d_in[1];
  const float* query = (const float*)d_in[2];
  const float* Wq = (const float*)d_in[3];
  const float* bq = (const float*)d_in[4];
  const float* Wk = (const float*)d_in[5];
  const float* bk = (const float*)d_in[6];
  const float* Wv = (const float*)d_in[7];
  const float* bv = (const float*)d_in[8];
  const float* Wo = (const float*)d_in[9];
  const float* bo = (const float*)d_in[10];
  float* out = (float*)d_out;

  const size_t BLD = (size_t)Bc * Lc * Dc;   // 8,388,608
  const size_t DD  = (size_t)Dc * Dc;        // 1,048,576

  char* ws = (char*)d_ws;
  size_t off = 0;
  auto alloc = [&](size_t bytes) -> void* {
    void* p = ws + off;
    off += (bytes + 255) & ~(size_t)255;
    return p;
  };
  unsigned short* qb  = (unsigned short*)alloc(BLD * 2);  // query bf16
  unsigned short* kb  = (unsigned short*)alloc(BLD * 2);  // key bf16
  unsigned short* vb  = (unsigned short*)alloc(BLD * 2);  // value bf16
  unsigned short* wqt = (unsigned short*)alloc(DD * 2);   // Wq^T bf16
  unsigned short* wkt = (unsigned short*)alloc(DD * 2);
  unsigned short* wvt = (unsigned short*)alloc(DD * 2);
  unsigned short* wot = (unsigned short*)alloc(DD * 2);
  unsigned short* Qp  = (unsigned short*)alloc(BLD * 2);  // projected Q bf16
  unsigned short* Kp  = (unsigned short*)alloc(BLD * 2);
  unsigned short* Vp  = (unsigned short*)alloc(BLD * 2);
  unsigned short* Vtr = (unsigned short*)alloc(BLD * 2);  // V^T per batch [B][D][L]
  unsigned short* Cx  = (unsigned short*)alloc(BLD * 2);  // attention output bf16

  auto cast = [&](const float* s, unsigned short* d, size_t n) {
    int n4 = (int)(n / 4);
    cast_bf16_kernel<<<(n4 + 255) / 256, 256, 0, stream>>>(s, d, n4);
  };
  cast(query, qb, BLD);
  cast(key,   kb, BLD);
  cast(value, vb, BLD);

  // weights: cast + transpose (f32 [K][N] -> bf16 [N][K])
  {
    dim3 blk(32, 8);
    dim3 grd(Dc / 32, Dc / 32);
    castT_bf16_kernel<<<grd, blk, 0, stream>>>(Wq, wqt, Dc, Dc);
    castT_bf16_kernel<<<grd, blk, 0, stream>>>(Wk, wkt, Dc, Dc);
    castT_bf16_kernel<<<grd, blk, 0, stream>>>(Wv, wvt, Dc, Dc);
    castT_bf16_kernel<<<grd, blk, 0, stream>>>(Wo, wot, Dc, Dc);
  }

  const int M = Bc * Lc;                                 // 8192
  const int gemmBlocks = (M / 256) * (Dc / 64);          // 512
  gemm_bias_kernel<<<gemmBlocks, 256, 0, stream>>>(qb, wqt, bq, Qp, nullptr, M, Dc, Dc);
  gemm_bias_kernel<<<gemmBlocks, 256, 0, stream>>>(kb, wkt, bk, Kp, nullptr, M, Dc, Dc);
  gemm_bias_kernel<<<gemmBlocks, 256, 0, stream>>>(vb, wvt, bv, Vp, nullptr, M, Dc, Dc);

  // V per-batch transpose: [L][D] -> [D][L]
  {
    dim3 blk(32, 8);
    dim3 grd(Dc / 32, Lc / 32, Bc);
    transpose_bf16_kernel<<<grd, blk, 0, stream>>>(Vp, Vtr, Lc, Dc);
  }

  const int attnBlocks = Bc * Hc * (Lc / 16) / 8;        // 1024
  attn_kernel<<<attnBlocks, 256, 0, stream>>>(Qp, Kp, Vtr, Cx);

  gemm_bias_kernel<<<gemmBlocks, 256, 0, stream>>>(Cx, wot, bo, nullptr, out, M, Dc, Dc);
}
